// Seq2SlateTransformerModel_30872224923760
// MI455X (gfx1250) — matrix operand order repaired and checked by reading, not compiled, not run on hardware
//
#include <hip/hip_runtime.h>

typedef __attribute__((ext_vector_type(8)))  _Float16 v8h;
typedef __attribute__((ext_vector_type(16))) _Float16 v16h;
typedef __attribute__((ext_vector_type(8)))  float    v8f;

#define F_TRANSB 1
#define F_RELU   2
#define F_RESID  4
#define F_OUTF32 8

#define BK   32
#define APAD 8

#if defined(__gfx1250__) && __has_builtin(__builtin_amdgcn_global_load_async_to_lds_b128) && \
    __has_builtin(__builtin_amdgcn_s_wait_asynccnt)
#define USE_ASYNC_LDS 1
#else
#define USE_ASYNC_LDS 0
#endif

// Builtin takes (v4i addrspace(1)* gsrc, v4i addrspace(3)* lds, imm, imm)
typedef int gcc_v4i __attribute__((vector_size(16)));
typedef __attribute__((address_space(1))) gcc_v4i gl_v4i;
typedef __attribute__((address_space(3))) gcc_v4i ld_v4i;

struct GemmP {
    const _Float16* A;
    const _Float16* Bmat;
    const float*    bias;
    const _Float16* resid;
    void*           C;
    int M, N, K;
    int lda, ldb, ldc;
    long long sA1, sA2, sB1, sB2, sC1, sC2;
    int sBias2;
    int Z2;
    float alpha;
    int flags;
};

__device__ inline v8h v8h_zero() {
    v8h v;
#pragma unroll
    for (int i = 0; i < 8; ++i) v[i] = (_Float16)0.f;
    return v;
}

// Copy one aligned 16B chunk global->LDS (async path when available).
__device__ inline void stage16B(const _Float16* gsrc, _Float16* lds_dst) {
#if USE_ASYNC_LDS
    __builtin_amdgcn_global_load_async_to_lds_b128((gl_v4i*)gsrc, (ld_v4i*)lds_dst,
                                                   0, 0);
#else
    *(v8h*)lds_dst = *(const v8h*)gsrc;
#endif
}

// Tiled WMMA GEMM: C[z] = (A[z] * B[z] + bias)*alpha (+resid) with opt relu.
// A row-major [M][K] f16.  B row-major [K][N] f16 (or [N][K] when F_TRANSB).
// LDS layout: As[M][K], Bs[N][K] so fragments load as 16B ds_load_b128 pairs.
template <int WROWS, int WCOLS, int WMT, int WNT>
__global__ __launch_bounds__(WROWS* WCOLS * 32) void gemm_wmma(GemmP p) {
    constexpr int BM = WROWS * WMT * 16;
    constexpr int BN = WCOLS * WNT * 16;
    constexpr int NT = WROWS * WCOLS * 32;

    __shared__ __align__(16) _Float16 As[BM][BK + APAD];
    __shared__ __align__(16) _Float16 Bs[BN][BK + APAD];

    const int tid  = threadIdx.x;
    const int wid  = tid >> 5;
    const int lane = tid & 31;
    const int wrow = wid / WCOLS;
    const int wcol = wid % WCOLS;
    const int lane_m  = lane & 15;
    const int lane_hi = lane >> 4;

    const int z  = blockIdx.z;
    const int z1 = z / p.Z2;
    const int z2 = z % p.Z2;
    const int m0 = blockIdx.y * BM;
    const int n0 = blockIdx.x * BN;

    const _Float16* A = p.A    + (long long)z1 * p.sA1 + (long long)z2 * p.sA2;
    const _Float16* B = p.Bmat + (long long)z1 * p.sB1 + (long long)z2 * p.sB2;

    v8f acc[WMT][WNT];
#pragma unroll
    for (int i = 0; i < WMT; ++i)
#pragma unroll
        for (int j = 0; j < WNT; ++j)
#pragma unroll
            for (int r = 0; r < 8; ++r) acc[i][j][r] = 0.f;

    for (int k0 = 0; k0 < p.K; k0 += BK) {
        const bool fullk = (k0 + BK) <= p.K;
        // ---- stage A tile (async/vector; rows zero-filled when OOB) ----
        for (int c = tid; c < BM * 4; c += NT) {
            const int row = c >> 2;
            const int kc  = (c & 3) * 8;
            const int gm  = m0 + row;
            if (gm < p.M && fullk) {
                stage16B(A + (long long)gm * p.lda + k0 + kc, &As[row][kc]);
            } else {
                v8h val = v8h_zero();
                if (gm < p.M) {
#pragma unroll
                    for (int e = 0; e < 8; ++e) {
                        const int kk = k0 + kc + e;
                        if (kk < p.K) val[e] = A[(long long)gm * p.lda + kk];
                    }
                }
                *(v8h*)&As[row][kc] = val;
            }
        }
        // ---- stage B tile ----
        if (p.flags & F_TRANSB) {
            for (int c = tid; c < BN * 4; c += NT) {
                const int row = c >> 2;
                const int kc  = (c & 3) * 8;
                const int gn  = n0 + row;
                if (gn < p.N && fullk) {
                    stage16B(B + (long long)gn * p.ldb + k0 + kc, &Bs[row][kc]);
                } else {
                    v8h val = v8h_zero();
                    if (gn < p.N) {
#pragma unroll
                        for (int e = 0; e < 8; ++e) {
                            const int kk = k0 + kc + e;
                            if (kk < p.K) val[e] = B[(long long)gn * p.ldb + kk];
                        }
                    }
                    *(v8h*)&Bs[row][kc] = val;
                }
            }
        } else {
            for (int i = tid; i < BN * BK; i += NT) {
                const int n = i % BN;
                const int k = i / BN;
                const int gn = n0 + n, gk = k0 + k;
                _Float16 v = (gn < p.N && gk < p.K) ? B[(long long)gk * p.ldb + gn]
                                                    : (_Float16)0.f;
                Bs[n][k] = v;
            }
        }
        {   // prefetch next A tile into caches (global_prefetch_b8)
            const int prow = m0 + (tid & (BM - 1));
            if (k0 + BK < p.K && prow < p.M)
                __builtin_prefetch(A + (long long)prow * p.lda + k0 + BK, 0, 3);
        }
#if USE_ASYNC_LDS
        __builtin_amdgcn_s_wait_asynccnt(0);
#endif
        __syncthreads();

        // ---- fragment loads + WMMA ----
        v16h af[WMT], bf[WNT];
#pragma unroll
        for (int i = 0; i < WMT; ++i) {
            const int row = wrow * WMT * 16 + i * 16 + lane_m;
            const int kh  = lane_hi * 8;  // lanes16-31 start at K=8 per ISA layout
            ((v8h*)&af[i])[0] = *(const v8h*)&As[row][kh];        // K kh..kh+7
            ((v8h*)&af[i])[1] = *(const v8h*)&As[row][kh + 16];   // K kh+16..kh+23
        }
#pragma unroll
        for (int j = 0; j < WNT; ++j) {
            const int col = wcol * WNT * 16 + j * 16 + lane_m;
            const int kb  = lane_hi * 8;  // lanes16-31 start at K=8 per ISA layout
            ((v8h*)&bf[j])[0] = *(const v8h*)&Bs[col][kb];
            ((v8h*)&bf[j])[1] = *(const v8h*)&Bs[col][kb + 16];
        }
#pragma unroll
        for (int i = 0; i < WMT; ++i)
#pragma unroll
            for (int j = 0; j < WNT; ++j)
                acc[i][j] = __builtin_amdgcn_wmma_f32_16x16x32_f16(
                    false, af[i], false, bf[j], (short)0, acc[i][j], false, false);
        __syncthreads();
    }

    // ---- epilogue ----
    const long long cz = (long long)z1 * p.sC1 + (long long)z2 * p.sC2;
#pragma unroll
    for (int i = 0; i < WMT; ++i) {
#pragma unroll
        for (int j = 0; j < WNT; ++j) {
            const int n = n0 + wcol * WNT * 16 + j * 16 + lane_m;
            if (n >= p.N) continue;
            const float bias = p.bias ? p.bias[(long long)z2 * p.sBias2 + n] : 0.f;
#pragma unroll
            for (int r = 0; r < 8; ++r) {
                const int m = m0 + wrow * WMT * 16 + i * 16 + lane_hi * 8 + r;
                if (m >= p.M) continue;
                float v = (acc[i][j][r] + bias) * p.alpha;
                const long long coff = cz + (long long)m * p.ldc + n;
                if (p.flags & F_RESID) v += (float)p.resid[coff];
                if (p.flags & F_RELU) v = v > 0.f ? v : 0.f;
                if (p.flags & F_OUTF32) ((float*)p.C)[coff] = v;
                else                    ((_Float16*)p.C)[coff] = (_Float16)v;
            }
        }
    }
}

__global__ void f32_to_f16(const float* __restrict__ s, _Float16* __restrict__ d,
                           long long n) {
    long long i = (long long)blockIdx.x * blockDim.x + threadIdx.x;
    if (i < n) d[i] = (_Float16)s[i];
}

// Batched transpose-convert: src (nmat of R x C, f32 row-major) ->
// dst (nmat of C x R, f16 row-major).  32x32 LDS tiles, block (32,8).
__global__ __launch_bounds__(256) void convT_kernel(const float* __restrict__ src,
                                                    _Float16* __restrict__ dst,
                                                    int R, int C) {
    __shared__ float t[32][33];
    const size_t m = blockIdx.z;
    const float* s = src + m * (size_t)R * C;
    _Float16*    d = dst + m * (size_t)R * C;
    const int c0 = blockIdx.x * 32, r0 = blockIdx.y * 32;
    for (int i = threadIdx.y; i < 32; i += 8) {
        const int r = r0 + i, c = c0 + threadIdx.x;
        t[i][threadIdx.x] = (r < R && c < C) ? s[(size_t)r * C + c] : 0.f;
    }
    __syncthreads();
    for (int i = threadIdx.y; i < 32; i += 8) {
        const int c = c0 + i, r = r0 + threadIdx.x;
        if (c < C && r < R) d[(size_t)c * R + r] = (_Float16)t[threadIdx.x][i];
    }
}

// LayerNorm over D=512, 256 threads, 2 elems/thread, fp32 stats.
__global__ __launch_bounds__(256) void ln_kernel(const _Float16* __restrict__ in,
                                                 const float* __restrict__ g,
                                                 const float* __restrict__ bb,
                                                 _Float16* __restrict__ out) {
    const long long row = blockIdx.x;
    const int tid = threadIdx.x;
    const _Float16* x = in + row * 512;
    const float a0 = (float)x[tid];
    const float a1 = (float)x[tid + 256];
    __shared__ float red[8];
    float s = a0 + a1;
#pragma unroll
    for (int o = 16; o; o >>= 1) s += __shfl_xor(s, o, 32);
    if ((tid & 31) == 0) red[tid >> 5] = s;
    __syncthreads();
    float tot = 0.f;
#pragma unroll
    for (int w = 0; w < 8; ++w) tot += red[w];
    const float mean = tot * (1.f / 512.f);
    const float d0 = a0 - mean, d1 = a1 - mean;
    float vs = d0 * d0 + d1 * d1;
    __syncthreads();
#pragma unroll
    for (int o = 16; o; o >>= 1) vs += __shfl_xor(vs, o, 32);
    if ((tid & 31) == 0) red[tid >> 5] = vs;
    __syncthreads();
    float var = 0.f;
#pragma unroll
    for (int w = 0; w < 8; ++w) var += red[w];
    const float inv = rsqrtf(var * (1.f / 512.f) + 1e-5f);
    out[row * 512 + tid]       = (_Float16)(d0 * inv * g[tid] + bb[tid]);
    out[row * 512 + tid + 256] = (_Float16)(d1 * inv * g[tid + 256] + bb[tid + 256]);
}

// Row softmax over LK (<=112) with optional causal mask; writes f16 P and fp32 back.
__global__ __launch_bounds__(128) void softmax_kernel(float* __restrict__ S,
                                                      _Float16* __restrict__ P,
                                                      int LQ, int LK, int ldr,
                                                      int causal) {
    const long long row = blockIdx.x;  // z*LQ + q
    const int q   = (int)(row % LQ);
    const int tid = threadIdx.x;
    float* srow = S + row * ldr;
    const bool valid = (tid < LK) && (!causal || tid <= q);
    const float v = valid ? srow[tid] : -1e30f;
    __shared__ float red[4];
    float m = v;
#pragma unroll
    for (int o = 16; o; o >>= 1) m = fmaxf(m, __shfl_xor(m, o, 32));
    if ((tid & 31) == 0) red[tid >> 5] = m;
    __syncthreads();
    m = fmaxf(fmaxf(red[0], red[1]), fmaxf(red[2], red[3]));
    const float e = valid ? expf(v - m) : 0.f;
    __syncthreads();
    float s = e;
#pragma unroll
    for (int o = 16; o; o >>= 1) s += __shfl_xor(s, o, 32);
    if ((tid & 31) == 0) red[tid >> 5] = s;
    __syncthreads();
    s = red[0] + red[1] + red[2] + red[3];
    const float r = valid ? e / s : 0.f;
    if (tid < ldr) {
        srow[tid] = r;
        P[row * ldr + tid] = (_Float16)r;
    }
}

// x[b,l,0:256] = state_e[b]; x[b,l,256:512] = cand_e[b,l]
__global__ void build_x(const _Float16* __restrict__ se, const _Float16* __restrict__ ce,
                        _Float16* __restrict__ x, int Ln) {
    const long long i = (long long)blockIdx.x * blockDim.x + threadIdx.x;
    const int d = (int)(i & 511);
    const long long t = i >> 9;
    const long long b = t / Ln;
    x[i] = (d < 256) ? se[b * 256 + d] : ce[t * 256 + (d - 256)];
}

// y = concat(state_e, cand_e[gather(idx)]) + pos
__global__ void build_y(const _Float16* __restrict__ se, const _Float16* __restrict__ ce,
                        const int* __restrict__ idx, const float* __restrict__ pos,
                        _Float16* __restrict__ y, int Lt, int Ls) {
    const long long i = (long long)blockIdx.x * blockDim.x + threadIdx.x;
    const int d = (int)(i & 511);
    const long long t = i >> 9;
    const long long b = t / Lt;
    const int tt = (int)(t % Lt);
    float v;
    if (d < 256) v = (float)se[b * 256 + d];
    else {
        const int s = idx[b * Lt + tt];
        v = (float)ce[(b * Ls + s) * 256 + (d - 256)];
    }
    v += pos[(long long)tt * 512 + d];
    y[i] = (_Float16)v;
}

// out[b,t,0:2]=0 ; out[b,t,2+s] = mean_h S[b,h,t,s]
__global__ __launch_bounds__(128) void avg_attn(const float* __restrict__ S,
                                                float* __restrict__ out, int Lt,
                                                int Ls, int ldr) {
    const long long row = blockIdx.x;  // b*Lt + t
    const int tid = threadIdx.x;
    if (tid >= Ls + 2) return;
    const long long b = row / Lt;
    const int t = (int)(row % Lt);
    float v = 0.f;
    if (tid >= 2) {
        const int s = tid - 2;
#pragma unroll
        for (int h = 0; h < 8; ++h)
            v += S[(((b * 8 + h) * Lt) + t) * (long long)ldr + s];
        v *= 0.125f;
    }
    out[row * (Ls + 2) + tid] = v;
}

static inline int cdiv(int a, int b) { return (a + b - 1) / b; }

static void run_gemm(hipStream_t st, bool big, const _Float16* A, const _Float16* B,
                     const float* bias, const _Float16* resid, void* C, int M, int N,
                     int K, int lda, int ldb, int ldc, float alpha, int flags,
                     int Z = 1, int Z2 = 1, long long sA1 = 0, long long sA2 = 0,
                     long long sB1 = 0, long long sB2 = 0, long long sC1 = 0,
                     long long sC2 = 0, int sBias2 = 0) {
    GemmP p;
    p.A = A; p.Bmat = B; p.bias = bias; p.resid = resid; p.C = C;
    p.M = M; p.N = N; p.K = K; p.lda = lda; p.ldb = ldb; p.ldc = ldc;
    p.sA1 = sA1; p.sA2 = sA2; p.sB1 = sB1; p.sB2 = sB2; p.sC1 = sC1; p.sC2 = sC2;
    p.sBias2 = sBias2; p.Z2 = Z2; p.alpha = alpha; p.flags = flags;
    if (big) {
        dim3 g(cdiv(N, 128), cdiv(M, 128), Z);
        gemm_wmma<4, 2, 2, 4><<<g, 256, 0, st>>>(p);
    } else {
        dim3 g(cdiv(N, 64), cdiv(M, 64), Z);
        gemm_wmma<2, 2, 2, 2><<<g, 128, 0, st>>>(p);
    }
}

extern "C" void kernel_launch(void* const* d_in, const int* in_sizes, int n_in,
                              void* d_out, int out_size, void* d_ws, size_t ws_size,
                              hipStream_t stream) {
    (void)in_sizes; (void)n_in; (void)out_size; (void)ws_size;
    const int Bn = 256, LS = 100, LT = 100, Hh = 8;
    const int LQP = 112;                       // padded score row (16B aligned f16/f32)
    const long long TOK = (long long)Bn * LS;  // 25600
    const long long QS  = TOK * 512;           // per-matrix stride in qkv buffer

    const float* in_state   = (const float*)d_in[0];
    const float* in_src     = (const float*)d_in[1];
    const int*   in_tgt_idx = (const int*)d_in[2];
    const float* p_state_W  = (const float*)d_in[3];
    const float* p_state_b  = (const float*)d_in[4];
    const float* p_cand_W   = (const float*)d_in[5];
    const float* p_cand_b   = (const float*)d_in[6];
    const float* p_pos      = (const float*)d_in[7];
    const float* p_encA_W   = (const float*)d_in[8];
    const float* p_encA_b   = (const float*)d_in[9];
    const float* p_encF1_W  = (const float*)d_in[10];
    const float* p_encF1_b  = (const float*)d_in[11];
    const float* p_encF2_W  = (const float*)d_in[12];
    const float* p_encF2_b  = (const float*)d_in[13];
    const float* p_encLN_g  = (const float*)d_in[14];
    const float* p_encLN_b  = (const float*)d_in[15];
    const float* p_encFin_g = (const float*)d_in[16];
    const float* p_encFin_b = (const float*)d_in[17];
    const float* p_decS_W   = (const float*)d_in[18];
    const float* p_decS_b   = (const float*)d_in[19];
    const float* p_decC_W   = (const float*)d_in[20];
    const float* p_decC_b   = (const float*)d_in[21];
    const float* p_decF1_W  = (const float*)d_in[22];
    const float* p_decF1_b  = (const float*)d_in[23];
    const float* p_decF2_W  = (const float*)d_in[24];
    const float* p_decF2_b  = (const float*)d_in[25];
    const float* p_decLN_g  = (const float*)d_in[26];
    const float* p_decLN_b  = (const float*)d_in[27];

    char* wsb = (char*)d_ws;
    size_t off = 0;
    auto alloc_h = [&](size_t n) -> _Float16* {
        _Float16* p = (_Float16*)(wsb + off);
        off += (n * 2 + 255) & ~(size_t)255;
        return p;
    };
    auto alloc_f = [&](size_t n) -> float* {
        float* p = (float*)(wsb + off);
        off += (n * 4 + 255) & ~(size_t)255;
        return p;
    };

    // All weight buffers hold the TRANSPOSED f16 weights ([N][K] layout).
    _Float16* stW16  = alloc_h(128 * 256);
    _Float16* caW16  = alloc_h(128 * 256);
    _Float16* eAW16  = alloc_h(2097152);
    _Float16* eF1W16 = alloc_h(2097152);
    _Float16* eF2W16 = alloc_h(2097152);
    _Float16* dSW16  = alloc_h(2097152);
    _Float16* dCW16  = alloc_h(2097152);
    _Float16* dF1W16 = alloc_h(2097152);
    _Float16* dF2W16 = alloc_h(2097152);
    _Float16* st16   = alloc_h(Bn * 128);
    _Float16* src16  = alloc_h(TOK * 128);
    _Float16* se16   = alloc_h(Bn * 256);
    _Float16* ce16   = alloc_h(TOK * 256);
    _Float16* x16    = alloc_h(QS);
    _Float16* h16    = alloc_h(QS);
    _Float16* big16  = alloc_h(TOK * 2048);  // qkv (3*QS) or FF hidden
    _Float16* ctx16  = alloc_h(QS);
    _Float16* tmp16  = alloc_h(QS);
    _Float16* y16    = alloc_h(QS);
    _Float16* mem16  = alloc_h(QS);
    float*    Sbuf   = alloc_f((size_t)Bn * Hh * LT * LQP);
    _Float16* P16    = alloc_h((size_t)Bn * Hh * LT * LQP);

    auto conv = [&](const float* s, _Float16* d, long long n) {
        f32_to_f16<<<(int)((n + 255) / 256), 256, 0, stream>>>(s, d, n);
    };
    auto convT = [&](const float* s, _Float16* d, int R, int C, int nmat) {
        dim3 g(cdiv(C, 32), cdiv(R, 32), nmat);
        convT_kernel<<<g, dim3(32, 8), 0, stream>>>(s, d, R, C);
    };
    conv(in_state, st16, Bn * 128);
    conv(in_src, src16, TOK * 128);
    convT(p_state_W, stW16, 128, 256, 1);
    convT(p_cand_W, caW16, 128, 256, 1);
    convT(p_encA_W, eAW16, 512, 512, 8);
    convT(p_encF1_W, eF1W16, 512, 2048, 2);
    convT(p_encF2_W, eF2W16, 2048, 512, 2);
    convT(p_decS_W, dSW16, 512, 512, 8);
    convT(p_decC_W, dCW16, 512, 512, 8);
    convT(p_decF1_W, dF1W16, 512, 2048, 2);
    convT(p_decF2_W, dF2W16, 2048, 512, 2);

    // ---- embeddings: (x @ W + b) * sqrt(256) ----
    run_gemm(stream, true, st16, stW16, p_state_b, nullptr, se16, Bn, 256, 128,
             128, 128, 256, 16.0f, F_TRANSB);
    run_gemm(stream, true, src16, caW16, p_cand_b, nullptr, ce16, (int)TOK, 256, 128,
             128, 128, 256, 16.0f, F_TRANSB);
    build_x<<<(int)(QS / 256), 256, 0, stream>>>(se16, ce16, x16, LS);

    const int NR = (int)TOK;           // token rows
    const int ZA = Bn * Hh;            // attention batches
    const int SMrows = ZA * LT;        // softmax rows
    const long long sTok = (long long)LS * 512;  // per-batch token stride
    const long long sSb = (long long)Hh * LT * LQP, sSh = (long long)LT * LQP;

    // ================= encoder =================
    for (int i = 0; i < 2; ++i) {
        const _Float16* Wa = eAW16 + (size_t)i * 1048576;
        const float* ba = p_encA_b + i * 2048;
        ln_kernel<<<NR, 256, 0, stream>>>(x16, p_encLN_g + (i * 2 + 0) * 512,
                                          p_encLN_b + (i * 2 + 0) * 512, h16);
        // q,k,v batched (z2 = 0..2)
        run_gemm(stream, true, h16, Wa, ba, nullptr, big16, NR, 512, 512, 512, 512,
                 512, 1.f, F_TRANSB, 3, 3, 0, 0, 0, 262144, 0, QS, 512);
        // scores = q k^T / 8
        run_gemm(stream, false, big16, big16 + QS, nullptr, nullptr, Sbuf, LT, LS,
                 64, 512, 512, LQP, 0.125f, F_TRANSB | F_OUTF32, ZA, Hh, sTok, 64,
                 sTok, 64, sSb, sSh);
        softmax_kernel<<<SMrows, 128, 0, stream>>>(Sbuf, P16, LT, LS, LQP, 0);
        // ctx = P v
        run_gemm(stream, false, P16, big16 + 2 * QS, nullptr, nullptr, ctx16, LT, 64,
                 LS, LQP, 512, 512, 1.f, 0, ZA, Hh, sSb, sSh, sTok, 64, sTok, 64);
        // x += ctx Wo + bo
        run_gemm(stream, true, ctx16, Wa + 3 * 262144, ba + 3 * 512, x16, x16, NR,
                 512, 512, 512, 512, 512, 1.f, F_TRANSB | F_RESID);
        // FF
        ln_kernel<<<NR, 256, 0, stream>>>(x16, p_encLN_g + (i * 2 + 1) * 512,
                                          p_encLN_b + (i * 2 + 1) * 512, h16);
        run_gemm(stream, true, h16, eF1W16 + (size_t)i * 1048576, p_encF1_b + i * 2048,
                 nullptr, big16, NR, 2048, 512, 512, 512, 2048, 1.f,
                 F_TRANSB | F_RELU);
        run_gemm(stream, true, big16, eF2W16 + (size_t)i * 1048576, p_encF2_b + i * 512,
                 x16, x16, NR, 512, 2048, 2048, 2048, 512, 1.f, F_TRANSB | F_RESID);
    }
    ln_kernel<<<NR, 256, 0, stream>>>(x16, p_encFin_g, p_encFin_b, mem16);

    // ---- decoder input ----
    build_y<<<(int)(QS / 256), 256, 0, stream>>>(se16, ce16, in_tgt_idx, p_pos, y16,
                                                 LT, LS);

    // ================= decoder =================
    for (int i = 0; i < 2; ++i) {
        const _Float16* Ws = dSW16 + (size_t)i * 1048576;
        const float* bs = p_decS_b + i * 2048;
        // self-attention (causal)
        run_gemm(stream, true, y16, Ws, bs, nullptr, big16, NR, 512, 512, 512, 512,
                 512, 1.f, F_TRANSB, 3, 3, 0, 0, 0, 262144, 0, QS, 512);
        run_gemm(stream, false, big16, big16 + QS, nullptr, nullptr, Sbuf, LT, LT,
                 64, 512, 512, LQP, 0.125f, F_TRANSB | F_OUTF32, ZA, Hh, sTok, 64,
                 sTok, 64, sSb, sSh);
        softmax_kernel<<<SMrows, 128, 0, stream>>>(Sbuf, P16, LT, LT, LQP, 1);
        run_gemm(stream, false, P16, big16 + 2 * QS, nullptr, nullptr, ctx16, LT, 64,
                 LT, LQP, 512, 512, 1.f, 0, ZA, Hh, sSb, sSh, sTok, 64, sTok, 64);
        run_gemm(stream, true, ctx16, Ws + 3 * 262144, bs + 3 * 512, y16, tmp16, NR,
                 512, 512, 512, 512, 512, 1.f, F_TRANSB | F_RESID);
        ln_kernel<<<NR, 256, 0, stream>>>(tmp16, p_decLN_g + (i * 3 + 0) * 512,
                                          p_decLN_b + (i * 3 + 0) * 512, y16);
        // cross-attention
        const _Float16* Wc = dCW16 + (size_t)i * 1048576;
        const float* bc = p_decC_b + i * 2048;
        run_gemm(stream, true, y16, Wc, bc, nullptr, big16, NR, 512, 512, 512, 512,
                 512, 1.f, F_TRANSB);                              // q
        run_gemm(stream, true, mem16, Wc + 262144, bc + 512, nullptr, big16 + QS, NR,
                 512, 512, 512, 512, 512, 1.f, F_TRANSB);          // k
        if (i == 0)
            run_gemm(stream, true, mem16, Wc + 2 * 262144, bc + 2 * 512, nullptr,
                     big16 + 2 * QS, NR, 512, 512, 512, 512, 512, 1.f,
                     F_TRANSB);                                    // v
        run_gemm(stream, false, big16, big16 + QS, nullptr, nullptr, Sbuf, LT, LS,
                 64, 512, 512, LQP, 0.125f, F_TRANSB | F_OUTF32, ZA, Hh, sTok, 64,
                 sTok, 64, sSb, sSh);
        softmax_kernel<<<SMrows, 128, 0, stream>>>(Sbuf, P16, LT, LS, LQP, 0);
        if (i == 0) {
            run_gemm(stream, false, P16, big16 + 2 * QS, nullptr, nullptr, ctx16, LT,
                     64, LS, LQP, 512, 512, 1.f, 0, ZA, Hh, sSb, sSh, sTok, 64,
                     sTok, 64);
            run_gemm(stream, true, ctx16, Wc + 3 * 262144, bc + 3 * 512, y16, tmp16,
                     NR, 512, 512, 512, 512, 512, 1.f, F_TRANSB | F_RESID);
            ln_kernel<<<NR, 256, 0, stream>>>(tmp16, p_decLN_g + (i * 3 + 1) * 512,
                                              p_decLN_b + (i * 3 + 1) * 512, y16);
            // FF
            run_gemm(stream, true, y16, dF1W16 + (size_t)i * 1048576,
                     p_decF1_b + i * 2048, nullptr, big16, NR, 2048, 512, 512, 512,
                     2048, 1.f, F_TRANSB | F_RELU);
            run_gemm(stream, true, big16, dF2W16 + (size_t)i * 1048576,
                     p_decF2_b + i * 512, y16, tmp16, NR, 512, 2048, 2048, 2048, 512,
                     1.f, F_TRANSB | F_RESID);
            ln_kernel<<<NR, 256, 0, stream>>>(tmp16, p_decLN_g + (i * 3 + 2) * 512,
                                              p_decLN_b + (i * 3 + 2) * 512, y16);
        } else {
            // final: head-averaged cross-attn weights, prepend 2 zero columns
            avg_attn<<<Bn * LT, 128, 0, stream>>>(Sbuf, (float*)d_out, LT, LS, LQP);
        }
    }
}
